// GCN_2396591751941
// MI455X (gfx1250) — matrix operand order repaired and checked by reading, not compiled, not run on hardware
//
#include <hip/hip_runtime.h>
#include <hip/hip_bf16.h>
#include <math.h>

typedef __attribute__((ext_vector_type(16))) _Float16 v16h;
typedef __attribute__((ext_vector_type(8)))  float    v8f;

#define N_FEAT 128
#define H_DIM 32
#define N_CLASSES 10
#define N_GRAPHS 64

// ---------------------------------------------------------------- utilities
__global__ void zero_kernel(float* __restrict__ p, int n) {
    int i = blockIdx.x * blockDim.x + threadIdx.x;
    if (i < n) p[i] = 0.0f;
}

// deg[dst] += 1 per edge (self-loop +1 folded into dis kernel)
__global__ void deg_kernel(const int* __restrict__ dst, float* __restrict__ deg, int E) {
    int e = blockIdx.x * blockDim.x + threadIdx.x;
    if (e < E) atomicAdd(&deg[dst[e]], 1.0f);
}

__global__ void dis_kernel(const float* __restrict__ deg, float* __restrict__ dis, int n) {
    int i = blockIdx.x * blockDim.x + threadIdx.x;
    if (i < n) dis[i] = rsqrtf(deg[i] + 1.0f);
}

// ---------------------------------------------------------------- weight pack
// Pre-pack W[K,32] (f32 row-major) into WMMA B-fragment f16 layout:
// out[((ks*2+tile)*32 + lane)*16 + i] = W[(ks*32 + (i>>3)*16 + (lane>>4)*8 + (i&7))*32 + tile*16 + (lane&15)]
// so the GEMM reads one contiguous 32B v16h per fragment per lane.
__global__ void pack_w(const float* __restrict__ W, _Float16* __restrict__ out, int K) {
    int t = blockIdx.x * blockDim.x + threadIdx.x;
    if (t >= K * H_DIM) return;
    int i    = t & 15;
    int l    = (t >> 4) & 31;
    int tile = (t >> 9) & 1;
    int ks   = t >> 10;
    out[t] = (_Float16)W[(ks * 32 + (i >> 3) * 16 + (l >> 4) * 8 + (i & 7)) * H_DIM + tile * 16 + (l & 15)];
}

// ---------------------------------------------------------------- WMMA GEMM
// C[M,32] = A[M,K] (f32 row-major) @ W[K,32] via v_wmma_f32_16x16x32_f16.
// One wave -> 16 rows x 32 cols; A loaded as float4, B as packed v16h.
__global__ void gemm_wmma(const float* __restrict__ A, const _Float16* __restrict__ Wp,
                          float* __restrict__ C, int M, int K) {
    const int wave = blockIdx.x * (blockDim.x >> 5) + (threadIdx.x >> 5);
    const int lane = threadIdx.x & 31;
    const int m0 = wave * 16;
    if (m0 >= M) return;                     // wave-uniform: EXEC stays all-1s

    const int half = lane >> 4;              // 0: lanes 0-15, 1: lanes 16-31
    const int idx  = lane & 15;              // row (for A) / column (for C)
    const float4* arow4 = (const float4*)(A + (size_t)(m0 + idx) * K);
    const v16h*   bp    = (const v16h*)Wp;

    v8f acc0 = {}; v8f acc1 = {};
    const int ksteps = K >> 5;
    for (int ks = 0; ks < ksteps; ++ks) {
        // A fragment: elems 0-7 <- K = ks*32 + half*8 + i ; elems 8-15 <- +16
        const int f4 = (ks << 3) + (half << 1);
        float4 q0 = arow4[f4];
        float4 q1 = arow4[f4 + 1];
        float4 q2 = arow4[f4 + 4];
        float4 q3 = arow4[f4 + 5];
        v16h a;
        a[0]=(_Float16)q0.x;  a[1]=(_Float16)q0.y;  a[2]=(_Float16)q0.z;  a[3]=(_Float16)q0.w;
        a[4]=(_Float16)q1.x;  a[5]=(_Float16)q1.y;  a[6]=(_Float16)q1.z;  a[7]=(_Float16)q1.w;
        a[8]=(_Float16)q2.x;  a[9]=(_Float16)q2.y;  a[10]=(_Float16)q2.z; a[11]=(_Float16)q2.w;
        a[12]=(_Float16)q3.x; a[13]=(_Float16)q3.y; a[14]=(_Float16)q3.z; a[15]=(_Float16)q3.w;

        v16h b0 = bp[(ks * 2 + 0) * 32 + lane];
        v16h b1 = bp[(ks * 2 + 1) * 32 + lane];

        acc0 = __builtin_amdgcn_wmma_f32_16x16x32_f16(false, a, false, b0,
                                                      (short)0, acc0, false, false);
        acc1 = __builtin_amdgcn_wmma_f32_16x16x32_f16(false, a, false, b1,
                                                      (short)0, acc1, false, false);
    }
    // C/D layout: VGPR r -> M = r + half*8, N = idx (+16 for acc1)
    float* crow = C + (size_t)(m0 + half * 8) * H_DIM + idx;
    #pragma unroll
    for (int r = 0; r < 8; ++r) {
        crow[(size_t)r * H_DIM]      = acc0[r];
        crow[(size_t)r * H_DIM + 16] = acc1[r];
    }
}

// ---------------------------------------------------------------- aggregation
// out[n,f] = dis[n]^2 * h[n,f] + b[f]   (self-loop term + bias init)
__global__ void agg_init(const float* __restrict__ h, const float* __restrict__ dis,
                         const float* __restrict__ b, float* __restrict__ out, int n) {
    int i = blockIdx.x * blockDim.x + threadIdx.x;
    if (i >= n * H_DIM) return;
    int node = i >> 5;
    int f = i & (H_DIM - 1);
    float ds = dis[node];
    out[i] = ds * ds * h[i] + b[f];
}

// out[dst,f] += dis[src]*dis[dst]*h[src,f]
// Each wave owns 32 edges: lane l loads edge e0+l's indices/norm once, then the
// wave sweeps the 32 edges via __shfl broadcast (lane = feature). 1 gather +
// 1 global_atomic_add_f32 per lane per edge.
__global__ void scatter_kernel(const float* __restrict__ h, const int* __restrict__ src,
                               const int* __restrict__ dst, const float* __restrict__ dis,
                               float* __restrict__ out, int E) {
    const int lane = threadIdx.x & 31;
    const int wave = (blockIdx.x * blockDim.x + threadIdx.x) >> 5;
    const int e0 = wave * 32;
    if (e0 >= E) return;                     // wave-uniform
    const int myE = min(32, E - e0);         // wave-uniform
    int s = 0, d = 0;
    float nrm = 0.0f;
    if (lane < myE) {
        s = src[e0 + lane];
        d = dst[e0 + lane];
        nrm = dis[s] * dis[d];
    }
    for (int j = 0; j < myE; ++j) {
        int   sj = __shfl(s, j);
        int   dj = __shfl(d, j);
        float nj = __shfl(nrm, j);
        atomicAdd(&out[(size_t)dj * H_DIM + lane], h[(size_t)sj * H_DIM + lane] * nj);
    }
}

__global__ void relu_kernel(float* __restrict__ p, int n) {
    int i = blockIdx.x * blockDim.x + threadIdx.x;
    if (i < n) p[i] = fmaxf(p[i], 0.0f);
}

// ---------------------------------------------------------------- pooling
// LDS partial sums per block (ds_add_f32), then one global atomic per cell.
__global__ void pool_kernel(const float* __restrict__ h, const int* __restrict__ batch,
                            float* __restrict__ psum, float* __restrict__ pcnt, int n) {
    __shared__ float ls[N_GRAPHS * H_DIM];
    __shared__ float lc[N_GRAPHS];
    for (int i = threadIdx.x; i < N_GRAPHS * H_DIM; i += blockDim.x) ls[i] = 0.0f;
    for (int i = threadIdx.x; i < N_GRAPHS; i += blockDim.x) lc[i] = 0.0f;
    __syncthreads();
    const int lane = threadIdx.x & 31;
    const int wave = threadIdx.x >> 5;
    const int wpb = blockDim.x >> 5;
    for (int node = blockIdx.x * wpb + wave; node < n; node += gridDim.x * wpb) {
        int g = batch[node];
        atomicAdd(&ls[g * H_DIM + lane], h[(size_t)node * H_DIM + lane]);
        if (lane == 0) atomicAdd(&lc[g], 1.0f);
    }
    __syncthreads();
    for (int i = threadIdx.x; i < N_GRAPHS * H_DIM; i += blockDim.x) atomicAdd(&psum[i], ls[i]);
    for (int i = threadIdx.x; i < N_GRAPHS; i += blockDim.x) atomicAdd(&pcnt[i], lc[i]);
}

// ---------------------------------------------------------------- head
// pooled = psum/max(cnt,1); logits = pooled@Wl + bl; softmax. 64 threads.
__global__ void head_kernel(const float* __restrict__ psum, const float* __restrict__ pcnt,
                            const float* __restrict__ Wl, const float* __restrict__ bl,
                            float* __restrict__ out) {
    int g = threadIdx.x;
    if (g >= N_GRAPHS) return;
    float inv = 1.0f / fmaxf(pcnt[g], 1.0f);
    float pooled[H_DIM];
    #pragma unroll
    for (int f = 0; f < H_DIM; ++f) pooled[f] = psum[g * H_DIM + f] * inv;
    float logits[N_CLASSES];
    float mx = -INFINITY;
    #pragma unroll
    for (int c = 0; c < N_CLASSES; ++c) {
        float acc = bl[c];
        #pragma unroll
        for (int f = 0; f < H_DIM; ++f) acc = fmaf(pooled[f], Wl[f * N_CLASSES + c], acc);
        logits[c] = acc;
        mx = fmaxf(mx, acc);
    }
    float s = 0.0f;
    #pragma unroll
    for (int c = 0; c < N_CLASSES; ++c) { logits[c] = expf(logits[c] - mx); s += logits[c]; }
    float rs = 1.0f / s;
    #pragma unroll
    for (int c = 0; c < N_CLASSES; ++c) out[g * N_CLASSES + c] = logits[c] * rs;
}

// ---------------------------------------------------------------- launcher
extern "C" void kernel_launch(void* const* d_in, const int* in_sizes, int n_in,
                              void* d_out, int out_size, void* d_ws, size_t ws_size,
                              hipStream_t stream) {
    const float* x     = (const float*)d_in[0];
    const int*   ei    = (const int*)d_in[1];
    const int*   batch = (const int*)d_in[2];
    const float* W1 = (const float*)d_in[3]; const float* b1 = (const float*)d_in[4];
    const float* W2 = (const float*)d_in[5]; const float* b2 = (const float*)d_in[6];
    const float* W3 = (const float*)d_in[7]; const float* b3 = (const float*)d_in[8];
    const float* Wl = (const float*)d_in[9]; const float* bl = (const float*)d_in[10];

    const int N = in_sizes[0] / N_FEAT;   // 100000
    const int E = in_sizes[1] / 2;        // 1600000
    const int* src = ei;
    const int* dst = ei + E;

    // workspace layout (~26.6 MB); all region sizes are 32B multiples,
    // keeping the packed-weight buffers aligned for v16h (b128) loads.
    float* deg  = (float*)d_ws;
    float* dis  = deg + N;
    float* hA   = dis + N;                        // GEMM outputs
    float* hB   = hA + (size_t)N * H_DIM;         // aggregated outputs
    float* psum = hB + (size_t)N * H_DIM;
    float* pcnt = psum + N_GRAPHS * H_DIM;
    _Float16* w1p = (_Float16*)(pcnt + N_GRAPHS); // 128*32 f16
    _Float16* w2p = w1p + N_FEAT * H_DIM;         // 32*32 f16
    _Float16* w3p = w2p + H_DIM * H_DIM;          // 32*32 f16

    const int T = 256;
    // degree + normalization + weight packing (independent, upfront)
    zero_kernel<<<(N + T - 1) / T, T, 0, stream>>>(deg, N);
    deg_kernel<<<(E + T - 1) / T, T, 0, stream>>>(dst, deg, E);
    dis_kernel<<<(N + T - 1) / T, T, 0, stream>>>(deg, dis, N);
    pack_w<<<(N_FEAT * H_DIM + T - 1) / T, T, 0, stream>>>(W1, w1p, N_FEAT);
    pack_w<<<(H_DIM * H_DIM + T - 1) / T, T, 0, stream>>>(W2, w2p, H_DIM);
    pack_w<<<(H_DIM * H_DIM + T - 1) / T, T, 0, stream>>>(W3, w3p, H_DIM);

    const int mtiles = (N + 15) / 16;             // 6250
    const int gemmBlocks = (mtiles + 3) / 4;      // 4 waves/block, 128 threads
    const int nf = N * H_DIM;
    const int eWaves = (E + 31) / 32;             // 32 edges per wave
    const int scatterBlocks = (eWaves + 7) / 8;   // 8 waves/block

    // layer 1 (K = 128)
    gemm_wmma<<<gemmBlocks, 128, 0, stream>>>(x, w1p, hA, N, N_FEAT);
    agg_init<<<(nf + T - 1) / T, T, 0, stream>>>(hA, dis, b1, hB, N);
    scatter_kernel<<<scatterBlocks, T, 0, stream>>>(hA, src, dst, dis, hB, E);
    relu_kernel<<<(nf + T - 1) / T, T, 0, stream>>>(hB, nf);

    // layer 2 (K = 32)
    gemm_wmma<<<gemmBlocks, 128, 0, stream>>>(hB, w2p, hA, N, H_DIM);
    agg_init<<<(nf + T - 1) / T, T, 0, stream>>>(hA, dis, b2, hB, N);
    scatter_kernel<<<scatterBlocks, T, 0, stream>>>(hA, src, dst, dis, hB, E);
    relu_kernel<<<(nf + T - 1) / T, T, 0, stream>>>(hB, nf);

    // layer 3 (K = 32, no relu)
    gemm_wmma<<<gemmBlocks, 128, 0, stream>>>(hB, w3p, hA, N, H_DIM);
    agg_init<<<(nf + T - 1) / T, T, 0, stream>>>(hA, dis, b3, hB, N);
    scatter_kernel<<<scatterBlocks, T, 0, stream>>>(hA, src, dst, dis, hB, E);

    // pooling + head
    zero_kernel<<<1, T, 0, stream>>>(psum, N_GRAPHS * H_DIM + N_GRAPHS);
    pool_kernel<<<512, T, 0, stream>>>(hB, batch, psum, pcnt, N);
    head_kernel<<<1, 64, 0, stream>>>(psum, pcnt, Wl, bl, (float*)d_out);
}
